// MultiHeadAttention_3178275799360
// MI455X (gfx1250) — matrix operand order repaired and checked by reading, not compiled, not run on hardware
//
#include <hip/hip_runtime.h>
#include <hip/hip_bf16.h>

// ---------------------------------------------------------------------------
// MHA for MI455X (gfx1250, wave32, WMMA bf16 -> f32, TDM double-buffered K)
// d_out = [ out: B*S*768 f32 | attn: B*H*S*S f32 ]
// ---------------------------------------------------------------------------

typedef __bf16 bf16_t;
typedef __attribute__((ext_vector_type(16))) __bf16 v16bf;
typedef __attribute__((ext_vector_type(8)))  float  v8f;
typedef __attribute__((ext_vector_type(4)))  unsigned int u32x4;
typedef __attribute__((ext_vector_type(8)))  int i32x8;
typedef __attribute__((ext_vector_type(4)))  int i32x4;

#define D_MODEL   768
#define NUM_HEADS 12
#define DEPTH     64
#define BATCH     2
#define SEQ       2048
#define M_TOT     (BATCH * SEQ)   // 4096
#define VSTR      80              // padded LDS stride (halves) for transposed V
#define LOG2E     1.44269504088896340736f

// Fragment union: WMMA bf16 operand viewed as b128 chunks (no half-repacking).
union FragBF { v16bf v; uint4 q[2]; unsigned u[8]; bf16_t h[16]; };

// Pack two f32 -> one dword of two bf16 (RNE), lo in [15:0], hi in [31:16].
__device__ __forceinline__ unsigned pkbf(float a, float b) {
  unsigned ua = __float_as_uint(a), ub = __float_as_uint(b);
  unsigned ra = (ua + 0x7FFFu + ((ua >> 16) & 1u)) >> 16;
  unsigned rb = (ub + 0x7FFFu + ((ub >> 16) & 1u)) & 0xFFFF0000u;
  return ra | rb;
}
__device__ __forceinline__ bf16_t f2bf(float f) {
  unsigned short s = (unsigned short)pkbf(f, 0.0f);
  union { unsigned short s; bf16_t b; } o; o.s = s;
  return o.b;
}

// ---------------------------------------------------------------------------
// Tensor Data Mover: 1-D contiguous copy global -> LDS (bytes % 8 == 0).
// ---------------------------------------------------------------------------
#if __has_builtin(__builtin_amdgcn_tensor_load_to_lds)
#define HAVE_TDM 1
#else
#define HAVE_TDM 0
#endif

__device__ __forceinline__ void tdm_load_1d(unsigned lds_byte_off,
                                            unsigned long long gaddr,
                                            unsigned bytes) {
#if HAVE_TDM
  const unsigned n = bytes >> 3;                  // 8-byte elements
  u32x4 g0;
  g0[0] = 1u;                                     // count=1, user descriptor
  g0[1] = lds_byte_off;                           // lds_addr
  g0[2] = (unsigned)gaddr;                        // global_addr[31:0]
  g0[3] = (unsigned)((gaddr >> 32) & 0x1FFFFFFull) | (2u << 30); // [56:32]|type=2
  i32x8 g1;
  g1[0] = (int)(3u << 16);                        // data_size = 8B
  g1[1] = (int)((n & 0xFFFFu) << 16);             // tensor_dim0[15:0]
  g1[2] = (int)((n >> 16) & 0xFFFFu);             // tensor_dim0[31:16]
  g1[3] = (int)((n & 0xFFFFu) << 16);             // tile_dim0 = n
  g1[4] = 0;                                      // tile_dim1/2 = 0 -> 1-D
  g1[5] = (int)n;                                 // tensor_dim0_stride
  g1[6] = 0;
  g1[7] = 0;
  i32x4 z4 = {0, 0, 0, 0};
#if __clang_major__ >= 23
  i32x8 z8 = {0, 0, 0, 0, 0, 0, 0, 0};
  __builtin_amdgcn_tensor_load_to_lds(g0, g1, z4, z4, z8, 0);
#else
  __builtin_amdgcn_tensor_load_to_lds(g0, g1, z4, z4, 0);
#endif
#else
  (void)lds_byte_off; (void)gaddr; (void)bytes;
#endif
}

// s_wait_tensorcnt needs a compile-time constant argument.
__device__ __forceinline__ void wait_tensor0() {
#if __has_builtin(__builtin_amdgcn_s_wait_tensorcnt)
  __builtin_amdgcn_s_wait_tensorcnt(0);
#else
  asm volatile("s_wait_tensorcnt 0" ::: "memory");
#endif
}
__device__ __forceinline__ void wait_tensor1() {
#if __has_builtin(__builtin_amdgcn_s_wait_tensorcnt)
  __builtin_amdgcn_s_wait_tensorcnt(1);
#else
  asm volatile("s_wait_tensorcnt 1" ::: "memory");
#endif
}

// ---------------------------------------------------------------------------
// One-time converts: activations f32 -> bf16; weights f32 -> bf16 transposed.
// ---------------------------------------------------------------------------
__global__ __launch_bounds__(256) void cvt_x_kernel(const float* __restrict__ X,
                                                    bf16_t* __restrict__ Xb) {
  int i = blockIdx.x * 256 + threadIdx.x;          // float4 index
  float4 x = ((const float4*)X)[i];
  uint2 o;
  o.x = pkbf(x.x, x.y);
  o.y = pkbf(x.z, x.w);
  ((uint2*)Xb)[i] = o;
}

__global__ __launch_bounds__(192) void cvt_wT_kernel(const float* __restrict__ W,
                                                     bf16_t* __restrict__ Wt) {
  int n = blockIdx.x;                              // output row (0..767)
  int k0 = threadIdx.x * 4;                        // 0..764
  float a = W[(size_t)(k0 + 0) * D_MODEL + n];
  float b = W[(size_t)(k0 + 1) * D_MODEL + n];
  float c = W[(size_t)(k0 + 2) * D_MODEL + n];
  float d = W[(size_t)(k0 + 3) * D_MODEL + n];
  uint2 o;
  o.x = pkbf(a, b);
  o.y = pkbf(c, d);
  *(uint2*)(Wt + (size_t)n * D_MODEL + k0) = o;
}

// ---------------------------------------------------------------------------
// Projection: Y[b,h,s,d] (bf16) = Xb[m,:] @ WtT[c,:] + bias[c],  c = h*64+d
// Both fragments: two contiguous b128 loads, no packing in the loop.
// ---------------------------------------------------------------------------
__global__ __launch_bounds__(256) void proj_kernel(const bf16_t* __restrict__ Xb,
                                                   const bf16_t* __restrict__ Wt,
                                                   const float* __restrict__ bias,
                                                   bf16_t* __restrict__ Y) {
  const int lane = threadIdx.x & 31;
  const int wave = threadIdx.x >> 5;
  const int tileId = blockIdx.x * 8 + wave;          // 12288 tiles
  const int TILES_N = D_MODEL / 16;                  // 48
  const int tile_m = tileId / TILES_N;
  const int tile_n = tileId % TILES_N;
  const int l16 = lane & 15;
  const int row = tile_m * 16 + l16;
  const int col = tile_n * 16 + l16;
  const int kbA = (lane < 16) ? 0 : 16;
  const int kbB = (lane < 16) ? 0 : 16;

  const bf16_t* ar0 = Xb + (size_t)row * D_MODEL + kbA;
  const bf16_t* br0 = Wt + (size_t)col * D_MODEL + kbB;
  v8f c = {};
#pragma unroll 2
  for (int k0 = 0; k0 < D_MODEL; k0 += 32) {
    FragBF a, b;
    a.q[0] = *(const uint4*)(ar0 + k0);
    a.q[1] = *(const uint4*)(ar0 + k0 + 8);
    b.q[0] = *(const uint4*)(br0 + k0);
    b.q[1] = *(const uint4*)(br0 + k0 + 8);
    c = __builtin_amdgcn_wmma_f32_16x16x32_bf16(false, a.v, false, b.v, (short)0,
                                                c, false, false);
  }
  const float bia = bias[col];
  const int h = col / DEPTH, d = col % DEPTH;
  const int rofs = (lane < 16) ? 0 : 8;
#pragma unroll
  for (int r = 0; r < 8; ++r) {
    int m = tile_m * 16 + r + rofs;
    int bb = m / SEQ, s = m % SEQ;
    Y[(((size_t)bb * NUM_HEADS + h) * SEQ + s) * DEPTH + d] = f2bf(c[r] + bia);
  }
}

// ---------------------------------------------------------------------------
// Fused attention, two-pass flash softmax. Block = 4 waves, wave = 16 q rows.
// K tiles: TDM double-buffered (prefetch depth 1, s_wait_tensorcnt 1).
// V staged transposed so PV B-fragments are contiguous b128 LDS loads.
// Softmax in exp2 domain (scale folded with log2 e).
// ---------------------------------------------------------------------------
__global__ __launch_bounds__(128) void attn_kernel(const bf16_t* __restrict__ Qh,
                                                   const bf16_t* __restrict__ Kh,
                                                   const bf16_t* __restrict__ Vh,
                                                   float* __restrict__ attn,
                                                   bf16_t* __restrict__ ctx) {
  __shared__ __align__(16) bf16_t Kt2[2][64 * DEPTH];    // 16 KB, double buffer
  __shared__ __align__(16) bf16_t VtT[DEPTH * VSTR];     // 10 KB, V transposed
  __shared__ __align__(16) bf16_t Ps[4][16 * 64];        // 8 KB, wave-private P

  const int lane = threadIdx.x & 31;
  const int wave = threadIdx.x >> 5;
  const int wv = __builtin_amdgcn_readfirstlane(threadIdx.x >> 5);  // uniform
  const int b = blockIdx.z, h = blockIdx.y;
  const int q0 = blockIdx.x * 64 + wave * 16;
  const int NT = SEQ / 64;                               // 32 key tiles

  const size_t bh = (size_t)b * NUM_HEADS + h;
  const bf16_t* Qb = Qh + bh * SEQ * DEPTH;
  const bf16_t* Kb = Kh + bh * SEQ * DEPTH;
  const bf16_t* Vb = Vh + bh * SEQ * DEPTH;
  float* attnb = attn + bh * SEQ * SEQ;

  const int l16 = lane & 15;
  const int kbA = (lane < 16) ? 0 : 16;
  const int kbB = (lane < 16) ? 0 : 16;
  const int rofs = (lane < 16) ? 0 : 8;
  const float scale2 = 0.125f * LOG2E;                   // 1/sqrt(64) * log2(e)
  unsigned koff[2];
  koff[0] = (unsigned)(uintptr_t)(void*)&Kt2[0][0];
  koff[1] = (unsigned)(uintptr_t)(void*)&Kt2[1][0];
  const unsigned long long kgbase =
      (unsigned long long)(uintptr_t)Kb + (unsigned)wv * 2048u;

  // Q fragments (two 16x32 A frags) -- contiguous b128 loads.
  FragBF aQ0, aQ1;
  {
    const bf16_t* qr = Qb + (size_t)(q0 + l16) * DEPTH;
    aQ0.q[0] = *(const uint4*)(qr + kbA);
    aQ0.q[1] = *(const uint4*)(qr + kbA + 8);
    aQ1.q[0] = *(const uint4*)(qr + 32 + kbA);
    aQ1.q[1] = *(const uint4*)(qr + 32 + kbA + 8);
  }

  float rmax[8], rsum[8];
#pragma unroll
  for (int r = 0; r < 8; ++r) { rmax[r] = -3.0e38f; rsum[r] = 0.0f; }

  // ---------------- pass 1: online softmax statistics (exp2 domain) --------
#if HAVE_TDM
  tdm_load_1d(koff[0] + (unsigned)wv * 2048u, kgbase, 2048u);   // prefetch t0
#endif
  for (int kt = 0; kt < NT; ++kt) {
    const bf16_t* Kt = &Kt2[kt & 1][0];
#if HAVE_TDM
    if (kt + 1 < NT) {
      tdm_load_1d(koff[(kt + 1) & 1] + (unsigned)wv * 2048u,
                  kgbase + (size_t)(kt + 1) * 64 * DEPTH * sizeof(bf16_t), 2048u);
      wait_tensor1();
    } else {
      wait_tensor0();
    }
#else
    {
      const uint4* src = (const uint4*)(Kb + (size_t)kt * 64 * DEPTH);
      uint4* dst = (uint4*)&Kt2[kt & 1][0];
      for (int i = threadIdx.x; i < 512; i += 128) dst[i] = src[i];
    }
#endif
    __syncthreads();
#pragma unroll
    for (int j = 0; j < 4; ++j) {
      const bf16_t* kr = &Kt[(j * 16 + l16) * DEPTH];
      FragBF b0, b1;
      b0.q[0] = *(const uint4*)(kr + kbB);
      b0.q[1] = *(const uint4*)(kr + kbB + 8);
      b1.q[0] = *(const uint4*)(kr + 32 + kbB);
      b1.q[1] = *(const uint4*)(kr + 32 + kbB + 8);
      v8f c = {};
      c = __builtin_amdgcn_wmma_f32_16x16x32_bf16(false, aQ0.v, false, b0.v, (short)0, c, false, false);
      c = __builtin_amdgcn_wmma_f32_16x16x32_bf16(false, aQ1.v, false, b1.v, (short)0, c, false, false);
#pragma unroll
      for (int r = 0; r < 8; ++r) {
        float v = c[r] * scale2;
        float m2 = fmaxf(rmax[r], v);
        rsum[r] = rsum[r] * exp2f(rmax[r] - m2) + exp2f(v - m2);
        rmax[r] = m2;
      }
    }
    __syncthreads();
  }
  // combine (max,sum) across the 16 lanes sharing each row
#pragma unroll
  for (int r = 0; r < 8; ++r) {
    float m = rmax[r], s = rsum[r];
#pragma unroll
    for (int off = 1; off < 16; off <<= 1) {
      float mo = __shfl_xor(m, off, 16);
      float so = __shfl_xor(s, off, 16);
      float mn = fmaxf(m, mo);
      s = s * exp2f(m - mn) + so * exp2f(mo - mn);
      m = mn;
    }
    rmax[r] = m;
    rsum[r] = 1.0f / s;                                  // reciprocal
  }

  // ---------------- pass 2: write attn + ctx accumulation ----------------
  v8f acc[4] = {v8f{}, v8f{}, v8f{}, v8f{}};
#if HAVE_TDM
  tdm_load_1d(koff[0] + (unsigned)wv * 2048u, kgbase, 2048u);   // prefetch t0
#endif
  for (int kt = 0; kt < NT; ++kt) {
    const bf16_t* Kt = &Kt2[kt & 1][0];
#if HAVE_TDM
    if (kt + 1 < NT)
      tdm_load_1d(koff[(kt + 1) & 1] + (unsigned)wv * 2048u,
                  kgbase + (size_t)(kt + 1) * 64 * DEPTH * sizeof(bf16_t), 2048u);
#else
    {
      const uint4* src = (const uint4*)(Kb + (size_t)kt * 64 * DEPTH);
      uint4* dst = (uint4*)&Kt2[kt & 1][0];
      for (int i = threadIdx.x; i < 512; i += 128) dst[i] = src[i];
    }
#endif
    // stage V transposed: VtT[d * VSTR + key] (overlaps the K DMA)
    {
      const uint4* srcV = (const uint4*)(Vb + (size_t)kt * 64 * DEPTH);
      for (int i = threadIdx.x; i < 512; i += 128) {
        union { uint4 q; bf16_t h[8]; } w;
        w.q = srcV[i];
        int key = i >> 3;
        int d0 = (i & 7) * 8;
#pragma unroll
        for (int e = 0; e < 8; ++e) VtT[(d0 + e) * VSTR + key] = w.h[e];
      }
    }
#if HAVE_TDM
    if (kt + 1 < NT) wait_tensor1(); else wait_tensor0();
#endif
    __syncthreads();

    // recompute logits, emit attn probabilities, stash P for restriping
#pragma unroll
    for (int j = 0; j < 4; ++j) {
      const bf16_t* kr = &Kt[(j * 16 + l16) * DEPTH];
      FragBF b0, b1;
      b0.q[0] = *(const uint4*)(kr + kbB);
      b0.q[1] = *(const uint4*)(kr + kbB + 8);
      b1.q[0] = *(const uint4*)(kr + 32 + kbB);
      b1.q[1] = *(const uint4*)(kr + 32 + kbB + 8);
      v8f c = {};
      c = __builtin_amdgcn_wmma_f32_16x16x32_bf16(false, aQ0.v, false, b0.v, (short)0, c, false, false);
      c = __builtin_amdgcn_wmma_f32_16x16x32_bf16(false, aQ1.v, false, b1.v, (short)0, c, false, false);
#pragma unroll
      for (int r = 0; r < 8; ++r) {
        float p = exp2f(c[r] * scale2 - rmax[r]) * rsum[r];
        int qr = q0 + r + rofs;
        attnb[(size_t)qr * SEQ + (size_t)kt * 64 + j * 16 + l16] = p;
        Ps[wave][(r + rofs) * 64 + j * 16 + l16] = f2bf(p);
      }
    }
    // wave-private LDS RAW: P stores must land before A-frag reads
    asm volatile("s_wait_dscnt 0" ::: "memory");

    FragBF aP0, aP1;
    {
      const bf16_t* pr = &Ps[wave][l16 * 64];
      aP0.q[0] = *(const uint4*)(pr + kbA);
      aP0.q[1] = *(const uint4*)(pr + kbA + 8);
      aP1.q[0] = *(const uint4*)(pr + 32 + kbA);
      aP1.q[1] = *(const uint4*)(pr + 32 + kbA + 8);
    }
#pragma unroll
    for (int t = 0; t < 4; ++t) {
      const bf16_t* vr = &VtT[(t * 16 + l16) * VSTR];
      FragBF b0, b1;
      b0.q[0] = *(const uint4*)(vr + kbB);
      b0.q[1] = *(const uint4*)(vr + kbB + 8);
      b1.q[0] = *(const uint4*)(vr + 32 + kbB);
      b1.q[1] = *(const uint4*)(vr + 32 + kbB + 8);
      acc[t] = __builtin_amdgcn_wmma_f32_16x16x32_bf16(false, aP0.v, false, b0.v, (short)0, acc[t], false, false);
      acc[t] = __builtin_amdgcn_wmma_f32_16x16x32_bf16(false, aP1.v, false, b1.v, (short)0, acc[t], false, false);
    }
    __syncthreads();
  }

  // ctx (bf16) in [B,S,768] concat-head layout for the output GEMM
#pragma unroll
  for (int t = 0; t < 4; ++t)
#pragma unroll
    for (int r = 0; r < 8; ++r) {
      int qr = q0 + r + rofs;
      ctx[((size_t)b * SEQ + qr) * D_MODEL + h * DEPTH + t * 16 + l16] = f2bf(acc[t][r]);
    }
}

// ---------------------------------------------------------------------------
// Output projection: out[m,n] (f32) = ctx(bf16)[m,:] @ WtT[n,:] + bo[n]
// ---------------------------------------------------------------------------
__global__ __launch_bounds__(256) void out_kernel(const bf16_t* __restrict__ A,
                                                  const bf16_t* __restrict__ Wt,
                                                  const float* __restrict__ bias,
                                                  float* __restrict__ out) {
  const int lane = threadIdx.x & 31;
  const int wave = threadIdx.x >> 5;
  const int tileId = blockIdx.x * 8 + wave;
  const int TILES_N = D_MODEL / 16;
  const int tile_m = tileId / TILES_N;
  const int tile_n = tileId % TILES_N;
  const int l16 = lane & 15;
  const int row = tile_m * 16 + l16;
  const int col = tile_n * 16 + l16;
  const int kbA = (lane < 16) ? 0 : 16;
  const int kbB = (lane < 16) ? 0 : 16;

  const bf16_t* ar0 = A + (size_t)row * D_MODEL + kbA;
  const bf16_t* br0 = Wt + (size_t)col * D_MODEL + kbB;
  v8f c = {};
#pragma unroll 2
  for (int k0 = 0; k0 < D_MODEL; k0 += 32) {
    FragBF a, b;
    a.q[0] = *(const uint4*)(ar0 + k0);
    a.q[1] = *(const uint4*)(ar0 + k0 + 8);
    b.q[0] = *(const uint4*)(br0 + k0);
    b.q[1] = *(const uint4*)(br0 + k0 + 8);
    c = __builtin_amdgcn_wmma_f32_16x16x32_bf16(false, a.v, false, b.v, (short)0,
                                                c, false, false);
  }
  const float bia = bias[col];
  const int rofs = (lane < 16) ? 0 : 8;
#pragma unroll
  for (int r = 0; r < 8; ++r) {
    int m = tile_m * 16 + r + rofs;
    out[(size_t)m * D_MODEL + col] = c[r] + bia;
  }
}

// ---------------------------------------------------------------------------
extern "C" void kernel_launch(void* const* d_in, const int* in_sizes, int n_in,
                              void* d_out, int out_size, void* d_ws, size_t ws_size,
                              hipStream_t stream) {
  (void)in_sizes; (void)n_in; (void)out_size; (void)ws_size;
  // setup_inputs order: v, k, q, wq, bq, wk, bk, wv, bv, wo, bo
  const float* v  = (const float*)d_in[0];
  const float* k  = (const float*)d_in[1];
  const float* q  = (const float*)d_in[2];
  const float* wq = (const float*)d_in[3];
  const float* bq = (const float*)d_in[4];
  const float* wk = (const float*)d_in[5];
  const float* bk = (const float*)d_in[6];
  const float* wv = (const float*)d_in[7];
  const float* bv = (const float*)d_in[8];
  const float* wo = (const float*)d_in[9];
  const float* bo = (const float*)d_in[10];

  const size_t headElems = (size_t)BATCH * NUM_HEADS * SEQ * DEPTH;  // 3.1M
  const size_t actElems  = (size_t)M_TOT * D_MODEL;                  // 3.1M
  bf16_t* qh  = (bf16_t*)d_ws;
  bf16_t* kh  = qh + headElems;
  bf16_t* vh  = kh + headElems;
  bf16_t* ctx = vh + headElems;            // [B,S,768] bf16
  bf16_t* xb  = ctx + actElems;            // reused activation buffer
  bf16_t* wT  = xb + actElems;             // reused transposed weight buffer

  float* out  = (float*)d_out;                              // [B,S,768]
  float* attn = out + (size_t)BATCH * SEQ * D_MODEL;        // [B,H,S,S]

  const int cvtBlocks  = (int)(actElems / 4 / 256);         // 3072
  const int projBlocks = (M_TOT / 16) * (D_MODEL / 16) / 8; // 1536

  // Q projection
  cvt_wT_kernel<<<D_MODEL, 192, 0, stream>>>(wq, wT);
  cvt_x_kernel<<<cvtBlocks, 256, 0, stream>>>(q, xb);
  proj_kernel<<<projBlocks, 256, 0, stream>>>(xb, wT, bq, qh);
  // K projection
  cvt_wT_kernel<<<D_MODEL, 192, 0, stream>>>(wk, wT);
  cvt_x_kernel<<<cvtBlocks, 256, 0, stream>>>(k, xb);
  proj_kernel<<<projBlocks, 256, 0, stream>>>(xb, wT, bk, kh);
  // V projection
  cvt_wT_kernel<<<D_MODEL, 192, 0, stream>>>(wv, wT);
  cvt_x_kernel<<<cvtBlocks, 256, 0, stream>>>(v, xb);
  proj_kernel<<<projBlocks, 256, 0, stream>>>(xb, wT, bv, vh);

  attn_kernel<<<dim3(SEQ / 64, NUM_HEADS, BATCH), 128, 0, stream>>>(qh, kh, vh,
                                                                    attn, ctx);

  cvt_wT_kernel<<<D_MODEL, 192, 0, stream>>>(wo, wT);
  out_kernel<<<projBlocks, 256, 0, stream>>>(ctx, wT, bo, out);
}
